// RSSM_6914897347204
// MI455X (gfx1250) — matrix operand order repaired and checked by reading, not compiled, not run on hardware
//
#include <hip/hip_runtime.h>

typedef __attribute__((ext_vector_type(16))) __bf16 v16bf;
typedef __attribute__((ext_vector_type(8)))  float  v8f;

#define T_STEPS 64
#define NB      256
#define E_DIM   1024
#define A_DIM   6
#define D_DIM   256
#define SD_DIM  1024
#define OUT_ROW 4352          // 4*SD + D
#define BLOCK_THREADS 512     // 16 wave32 -> 4 waves per SIMD32
#define WAVES 16

// ---------- helpers ----------
__device__ __forceinline__ unsigned short f2bf(float f) {
  unsigned int u = __float_as_uint(f);
  unsigned int r = u + 0x7FFFu + ((u >> 16) & 1u);   // round-to-nearest-even
  return (unsigned short)(r >> 16);
}
__device__ __forceinline__ float bf2f(unsigned short h) {
  return __uint_as_float(((unsigned int)h) << 16);
}
__device__ __forceinline__ float gumbel_hash(unsigned int idx) {
  unsigned int x = idx * 0x9E3779B1u + 0x85EBCA77u;
  x ^= x >> 16; x *= 0x85EBCA6Bu; x ^= x >> 13; x *= 0xC2B2AE35u; x ^= x >> 16;
  float u = (float)(x >> 8) * (1.0f / 16777216.0f);
  u = u * 0.9999998f + 1e-7f;                 // keep in (0,1)
  return -__logf(-__logf(u));
}
__device__ __forceinline__ float sigmoidf(float x) { return 1.0f / (1.0f + __expf(-x)); }

union FragU { v16bf v; uint4 q[2]; };

// ---------- weight pre-shuffle into WMMA B-fragment layout ----------
// dst element index i = ((kk*ntiles + nt)*32 + lane)*16 + e
// maps to  W[kk*32 + e + 16*(lane>>4)][nt*16 + (lane&15)]   (bf16, zero-padded K)
__global__ void shuffle_weight(const float* __restrict__ src, unsigned short* __restrict__ dst,
                               int Ksrc, int Kpad, int N) {
  int total  = Kpad * N;
  int ntiles = N >> 4;
  for (int i = blockIdx.x * blockDim.x + threadIdx.x; i < total;
       i += gridDim.x * blockDim.x) {
    int e    = i & 15;
    int lane = (i >> 4) & 31;
    int f    = i >> 9;
    int nt   = f % ntiles;
    int kk   = f / ntiles;
    int k    = kk * 32 + e + 16 * (lane >> 4);
    int n    = nt * 16 + (lane & 15);
    float v  = (k < Ksrc) ? src[(size_t)k * N + n] : 0.0f;
    dst[i] = f2bf(v);
  }
}

// ---------- one 16x16 output tile, K = 32*ksteps, A from LDS, B from shuffled global ----------
__device__ __forceinline__ v8f mm_tile(const unsigned short* inb, int aoff,
                                       const unsigned short* __restrict__ W,
                                       int ntilesW, int nt, int ksteps,
                                       float bias, int lane) {
  int m = lane & 15, hi = lane >> 4;
  v8f acc;
#pragma unroll
  for (int r = 0; r < 8; ++r) acc[r] = bias;
  const unsigned short* arow = inb + m * 1568 + aoff + 16 * hi;
  for (int kk = 0; kk < ksteps; ++kk) {
    FragU a, b;
    const uint4* ap = reinterpret_cast<const uint4*>(arow + kk * 32);
    a.q[0] = ap[0];   // elements 0..7  : K = k0 + 16*hi + 0..7
    a.q[1] = ap[1];   // elements 8..15 : K = k0 + 16*hi + 8..15
    const uint4* wp = reinterpret_cast<const uint4*>(
        W + (((size_t)(kk * ntilesW + nt)) * 32 + lane) * 16);
    b.q[0] = wp[0];
    b.q[1] = wp[1];
    acc = __builtin_amdgcn_wmma_f32_16x16x32_bf16(false, a.v, false, b.v,
                                                  (short)0, acc, false, false);
  }
  return acc;
}

// ---------- persistent RSSM kernel: 1 workgroup == 16 batch rows through all T steps ----------
__global__ __launch_bounds__(BLOCK_THREADS, 1)
void rssm_kernel(const float* __restrict__ embed, const float* __restrict__ action,
                 const float* __restrict__ isFirst,
                 const unsigned short* __restrict__ wImgIn,
                 const unsigned short* __restrict__ wIh,
                 const unsigned short* __restrict__ wHh,
                 const unsigned short* __restrict__ wImgOut,
                 const unsigned short* __restrict__ wObsOut,
                 const float* __restrict__ bImgIn, const float* __restrict__ bIh,
                 const float* __restrict__ bHh, const float* __restrict__ bImgOut,
                 const float* __restrict__ bObsOut,
                 float* __restrict__ out) {
  // inbuf row layout (bf16): [x 0:256 | deter 256:512 | stoch-or-emb 512:1536 | act 1536:1568]
  __shared__ unsigned short inbuf[16 * 1568];
  __shared__ unsigned short gbuf [16 * 1536];   // [gi r,z,n | gh r,z,n]
  __shared__ unsigned short lbuf [16 * 1024];   // logits (bf16) for sampling
  __shared__ float          dbuf [16 * 256];    // fp32 GRU carry

  const int tid  = threadIdx.x;
  const int lane = tid & 31;
  const int wave = tid >> 5;
  const int b0   = blockIdx.x * 16;
  const int mL   = lane & 15, hiL = lane >> 4;

  for (int i = tid; i < 16 * 1568; i += BLOCK_THREADS) inbuf[i] = 0;
  for (int i = tid; i < 16 * 256;  i += BLOCK_THREADS) dbuf[i]  = 0.0f;
  __syncthreads();

  for (int t = 0; t < T_STEPS; ++t) {
    // ---- load action for this step ----
    if (tid < 16 * A_DIM) {
      int m = tid / A_DIM, a = tid % A_DIM;
      inbuf[m * 1568 + 1536 + a] =
          f2bf(action[((size_t)(b0 + m) * T_STEPS + t) * A_DIM + a]);
    }
    __syncthreads();

    // ---- P2: x = [stoch|act] @ W_img_in + b   (writes inbuf[0:256], reads [512:1568]) ----
    for (int nt = wave; nt < 16; nt += WAVES) {
      float bias = bImgIn[nt * 16 + mL];
      v8f acc = mm_tile(inbuf, 512, wImgIn, 16, nt, 33, bias, lane);
      int col = nt * 16 + mL;
#pragma unroll
      for (int r = 0; r < 8; ++r)
        inbuf[(r + 8 * hiL) * 1568 + col] = f2bf(acc[r]);
    }
    __syncthreads();

    // ---- P3: gi = x@W_ih + b_ih ; gh = deter@W_hh + b_hh ----
    for (int tile = wave; tile < 96; tile += WAVES) {
      bool isGh = tile >= 48;
      int  nt   = isGh ? tile - 48 : tile;
      const unsigned short* W  = isGh ? wHh : wIh;
      const float*          bb = isGh ? bHh : bIh;
      int aoff = isGh ? 256 : 0;
      float bias = bb[nt * 16 + mL];
      v8f acc = mm_tile(inbuf, aoff, W, 48, nt, 8, bias, lane);
      int col = (isGh ? 768 : 0) + nt * 16 + mL;
#pragma unroll
      for (int r = 0; r < 8; ++r)
        gbuf[(r + 8 * hiL) * 1536 + col] = f2bf(acc[r]);
    }
    __syncthreads();

    // ---- P4: GRU combine (fp32) + stage emb into inbuf[512:1536] ----
    for (int i = tid; i < 16 * 256; i += BLOCK_THREADS) {
      int m = i >> 8, n = i & 255;
      const unsigned short* g = gbuf + m * 1536;
      float ir = bf2f(g[n]),        iz = bf2f(g[256 + n]),  inn = bf2f(g[512 + n]);
      float hr = bf2f(g[768 + n]),  hz = bf2f(g[1024 + n]), hn  = bf2f(g[1280 + n]);
      float r  = sigmoidf(ir + hr);
      float z  = sigmoidf(iz + hz);
      float nn = tanhf(inn + r * hn);
      float h  = (1.0f - z) * nn + z * dbuf[i];
      dbuf[i] = h;
      inbuf[m * 1568 + 256 + n] = f2bf(h);
      out[((size_t)(b0 + m) * T_STEPS + t) * OUT_ROW + 2048 + n] = h;
    }
    for (int i = tid; i < 16 * 1024; i += BLOCK_THREADS) {
      int m = i >> 10, n = i & 1023;
      inbuf[m * 1568 + 512 + n] =
          f2bf(embed[((size_t)(b0 + m) * T_STEPS + t) * E_DIM + n]);
    }
    __syncthreads();

    // ---- P5: prior_logits = deter @ W_img_out + b ----
    for (int nt = wave; nt < 64; nt += WAVES) {
      float bias = bImgOut[nt * 16 + mL];
      v8f acc = mm_tile(inbuf, 256, wImgOut, 64, nt, 8, bias, lane);
      int col = nt * 16 + mL;
#pragma unroll
      for (int r = 0; r < 8; ++r) {
        int row = r + 8 * hiL;
        out[((size_t)(b0 + row) * T_STEPS + t) * OUT_ROW + 2304 + col] = acc[r];
        lbuf[row * 1024 + col] = f2bf(acc[r]);
      }
    }
    __syncthreads();

    // ---- P6: prior straight-through sample (draw 0) ----
    for (int p = tid; p < 512; p += BLOCK_THREADS) {
      int m = p >> 5, s = p & 31;
      int b = b0 + m;
      float l[32];
      float maxl = -1e30f, best = -1e30f; int arg = 0;
      unsigned int nbase = (unsigned int)(((0 * T_STEPS + t) * NB + b) * 1024 + s * 32);
#pragma unroll
      for (int c = 0; c < 32; ++c) {
        float lv = bf2f(lbuf[m * 1024 + s * 32 + c]);
        l[c] = lv;
        maxl = fmaxf(maxl, lv);
        float pert = lv + gumbel_hash(nbase + c);
        if (pert > best) { best = pert; arg = c; }
      }
      float sum = 0.0f;
#pragma unroll
      for (int c = 0; c < 32; ++c) sum += __expf(l[c] - maxl);
      float inv = 1.0f / sum;
      size_t base = ((size_t)b * T_STEPS + t) * OUT_ROW + 3328 + (size_t)s * 32;
#pragma unroll
      for (int c = 0; c < 32; ++c) {
        float prob = __expf(l[c] - maxl) * inv;
        out[base + c] = ((c == arg ? 1.0f : 0.0f) + prob) - prob;
      }
    }
    __syncthreads();

    // ---- P7: post_logits = [deter|emb] @ W_obs_out + b   (K = 1280) ----
    for (int nt = wave; nt < 64; nt += WAVES) {
      float bias = bObsOut[nt * 16 + mL];
      v8f acc = mm_tile(inbuf, 256, wObsOut, 64, nt, 40, bias, lane);
      int col = nt * 16 + mL;
#pragma unroll
      for (int r = 0; r < 8; ++r) {
        int row = r + 8 * hiL;
        out[((size_t)(b0 + row) * T_STEPS + t) * OUT_ROW + col] = acc[r];
        lbuf[row * 1024 + col] = f2bf(acc[r]);
      }
    }
    __syncthreads();

    // ---- P8: post sample (draw 1) + next-step state masking ----
    for (int p = tid; p < 512; p += BLOCK_THREADS) {
      int m = p >> 5, s = p & 31;
      int b = b0 + m;
      float masknext = (t + 1 < T_STEPS)
                         ? (1.0f - isFirst[(size_t)b * T_STEPS + t + 1]) : 1.0f;
      float l[32];
      float maxl = -1e30f, best = -1e30f; int arg = 0;
      unsigned int nbase = (unsigned int)(((1 * T_STEPS + t) * NB + b) * 1024 + s * 32);
#pragma unroll
      for (int c = 0; c < 32; ++c) {
        float lv = bf2f(lbuf[m * 1024 + s * 32 + c]);
        l[c] = lv;
        maxl = fmaxf(maxl, lv);
        float pert = lv + gumbel_hash(nbase + c);
        if (pert > best) { best = pert; arg = c; }
      }
      float sum = 0.0f;
#pragma unroll
      for (int c = 0; c < 32; ++c) sum += __expf(l[c] - maxl);
      float inv = 1.0f / sum;
      size_t base = ((size_t)b * T_STEPS + t) * OUT_ROW + 1024 + (size_t)s * 32;
#pragma unroll
      for (int c = 0; c < 32; ++c) {
        float prob = __expf(l[c] - maxl) * inv;
        float st = ((c == arg ? 1.0f : 0.0f) + prob) - prob;
        out[base + c] = st;
        inbuf[m * 1568 + 512 + s * 32 + c] = f2bf(st * masknext);  // stoch for t+1
      }
    }
    for (int i = tid; i < 16 * 256; i += BLOCK_THREADS) {           // mask deter for t+1
      int m = i >> 8, n = i & 255;
      float masknext = (t + 1 < T_STEPS)
                         ? (1.0f - isFirst[(size_t)(b0 + m) * T_STEPS + t + 1]) : 1.0f;
      float h = dbuf[i] * masknext;
      dbuf[i] = h;
      inbuf[m * 1568 + 256 + n] = f2bf(h);
    }
    __syncthreads();
  }
}

// ---------- launcher ----------
extern "C" void kernel_launch(void* const* d_in, const int* in_sizes, int n_in,
                              void* d_out, int out_size, void* d_ws, size_t ws_size,
                              hipStream_t stream) {
  const float* embed     = (const float*)d_in[0];
  const float* action    = (const float*)d_in[1];
  const float* isFirst   = (const float*)d_in[2];
  const float* W_img_in  = (const float*)d_in[3];
  const float* b_img_in  = (const float*)d_in[4];
  const float* W_ih      = (const float*)d_in[5];
  const float* b_ih      = (const float*)d_in[6];
  const float* W_hh      = (const float*)d_in[7];
  const float* b_hh      = (const float*)d_in[8];
  const float* W_img_out = (const float*)d_in[9];
  const float* b_img_out = (const float*)d_in[10];
  const float* W_obs_out = (const float*)d_in[11];
  const float* b_obs_out = (const float*)d_in[12];
  float* out = (float*)d_out;

  unsigned short* ws = (unsigned short*)d_ws;
  unsigned short* wImgIn  = ws;            size_t off = (size_t)1056 * 256;
  unsigned short* wIh     = ws + off;      off += (size_t)256 * 768;
  unsigned short* wHh     = ws + off;      off += (size_t)256 * 768;
  unsigned short* wImgOut = ws + off;      off += (size_t)256 * 1024;
  unsigned short* wObsOut = ws + off;      off += (size_t)1280 * 1024;

  shuffle_weight<<<512, 256, 0, stream>>>(W_img_in,  wImgIn,  1030, 1056, 256);
  shuffle_weight<<<512, 256, 0, stream>>>(W_ih,      wIh,      256,  256, 768);
  shuffle_weight<<<512, 256, 0, stream>>>(W_hh,      wHh,      256,  256, 768);
  shuffle_weight<<<512, 256, 0, stream>>>(W_img_out, wImgOut,  256,  256, 1024);
  shuffle_weight<<<512, 256, 0, stream>>>(W_obs_out, wObsOut, 1280, 1280, 1024);

  rssm_kernel<<<16, BLOCK_THREADS, 0, stream>>>(
      embed, action, isFirst,
      wImgIn, wIh, wHh, wImgOut, wObsOut,
      b_img_in, b_ih, b_hh, b_img_out, b_obs_out, out);
}
